// LinearAttention_88012469830514
// MI455X (gfx1250) — compile-verified, matrix operand order checked by reading, not run on hardware
//
#include <hip/hip_runtime.h>
#include <hip/hip_bf16.h>

typedef __attribute__((ext_vector_type(16))) __bf16 v16bf;
typedef __attribute__((ext_vector_type(8)))  __bf16 v8bf;
typedef __attribute__((ext_vector_type(4)))  __bf16 v4bf;
typedef __attribute__((ext_vector_type(8)))  float  v8f;
typedef __attribute__((ext_vector_type(4)))  float  v4f;

#define B_ROWS  32768
#define DIM     2048
#define HEADS   8
#define DH      64
#define HIDDEN  512
#define NQKV    1536

#define KT1 (DIM / 32)      // 64 K-tiles for GEMM1
#define NT1 (NQKV / 16)     // 96 N-tiles for GEMM1
#define KT2 (HIDDEN / 32)   // 16 K-tiles for GEMM2
#define NT2 (DIM / 16)      // 128 N-tiles for GEMM2

#define XS_STRIDE 2056      // 2048 + 8 bf16 pad -> row advances 4 LDS banks
#define EPS_STRIDE 196      // 192 + 4 f32 pad
#define MS_STRIDE 520       // 512 + 8 bf16 pad

// ---- A fragment (16x32 bf16) from row-major LDS, per ISA 7.12.2 layout ----
// lane 0-15: M=lane, K = 0..7 & 16..23 ; lane 16-31: same M, K = 8..15 & 24..31
__device__ __forceinline__ v16bf load_a_frag(const __bf16* lds, int stride,
                                             int mbase, int kbase, int lane) {
    const int m  = lane & 15;
    const int kh = (lane >> 4) * 8;
    const __bf16* p = lds + (size_t)(mbase + m) * stride + kbase + kh;
    v8bf lo = *(const v8bf*)p;          // ds_load_b128
    v8bf hi = *(const v8bf*)(p + 16);   // ds_load_b128
    return __builtin_shufflevector(lo, hi, 0,1,2,3,4,5,6,7,8,9,10,11,12,13,14,15);
}

// ---- B fragment (32x16 bf16) from pre-swizzled weights: 1KB block / fragment,
// 32 contiguous bytes per lane -> 2x global_load_b128, fully coalesced ----
__device__ __forceinline__ v16bf load_b_frag(const __bf16* wsw, int ot, int kt,
                                             int Ktiles, int lane) {
    const __bf16* p = wsw + (((size_t)ot * Ktiles + kt) << 9) + (lane << 4);
    return *(const v16bf*)p;
}

// ---- CDNA5 async copy: 16B/lane global -> LDS, tracked by ASYNCcnt ----
// GVS mode: mem_addr = SGPR64 + VGPR32 offset; LDS dest per lane from VGPR.
__device__ __forceinline__ void async_copy_b128(unsigned lds_addr,
                                                const void* sbase, int voff) {
    asm volatile("global_load_async_to_lds_b128 %0, %1, %2"
                 :
                 : "v"(lds_addr), "v"(voff), "s"(sbase)
                 : "memory");
}

__device__ __forceinline__ void wait_async_zero() {
    asm volatile("s_wait_asynccnt 0x0" ::: "memory");
}

// ---- prepass: fp32 weights [Nout, K] -> bf16 fragment-swizzled layout ----
__global__ void la_swizzle_w_kernel(const float* __restrict__ w,
                                    __bf16* __restrict__ out,
                                    int Ktiles, int K, int total_frags) {
    int idx = blockIdx.x * 256 + threadIdx.x;
    if (idx >= total_frags * 32) return;
    int lane = idx & 31;
    int frag = idx >> 5;
    int kt = frag % Ktiles;
    int ot = frag / Ktiles;
    int n  = lane & 15;
    int kh = (lane >> 4) * 8;
    const float* src = w + (size_t)(ot * 16 + n) * K + kt * 32 + kh;
    v16bf o;
#pragma unroll
    for (int i = 0; i < 8; ++i) {
        o[i]     = (__bf16)src[i];        // K = kh .. kh+7
        o[i + 8] = (__bf16)src[16 + i];   // K = 16+kh .. 16+kh+7
    }
    *(v16bf*)(out + (size_t)frag * 512 + lane * 16) = o;
}

// ---- fused: qkv GEMM + per-head softmax attention -> mid[B,512] bf16 ----
__global__ __launch_bounds__(256, 1)
void la_qkv_attn_kernel(const float* __restrict__ x,
                        const __bf16* __restrict__ wqkv_sw,
                        __bf16* __restrict__ mid) {
    __shared__ __bf16 xs[64 * XS_STRIDE];   // 257 KB: 64-row x tile (bf16)
    __shared__ float  eps[64 * EPS_STRIDE]; // 49 KB: per-head q|k|v slab

    const int tid  = threadIdx.x;
    const int lane = tid & 31;
    const int wave = tid >> 5;
    const int rowbase = blockIdx.x * 64;

    // stage x tile -> bf16 LDS (needs fp32->bf16 convert, so VALU path)
    for (int t = tid; t < 64 * (DIM / 4); t += 256) {
        int r  = t >> 9;
        int c4 = t & 511;
        v4f f = *(const v4f*)(x + (size_t)(rowbase + r) * DIM + c4 * 4);
        v4bf h;
        h[0] = (__bf16)f[0]; h[1] = (__bf16)f[1];
        h[2] = (__bf16)f[2]; h[3] = (__bf16)f[3];
        *(v4bf*)&xs[r * XS_STRIDE + c4 * 4] = h;
    }
    __syncthreads();

    const int mtile = wave & 3;          // 4 M-strips of 16 rows
    const int jbase = (wave >> 2) * 6;   // 12 N-tiles (q4|k4|v4) split 2 ways

    for (int h = 0; h < HEADS; ++h) {
        v8f acc[6] = {};
        for (int kb = 0; kb < KT1; ++kb) {
            v16bf a = load_a_frag(xs, XS_STRIDE, mtile * 16, kb * 32, lane);
#pragma unroll
            for (int j2 = 0; j2 < 6; ++j2) {
                int j    = jbase + j2;
                int part = j >> 2;                  // 0=q 1=k 2=v
                int ot   = part * 32 + h * 4 + (j & 3);
                v16bf b = load_b_frag(wqkv_sw, ot, kb, KT1, lane);
                acc[j2] = __builtin_amdgcn_wmma_f32_16x16x32_bf16(
                    false, a, false, b, (short)0, acc[j2], false, false);
            }
        }

        __syncthreads();  // previous head's eps readers are done
#pragma unroll
        for (int j2 = 0; j2 < 6; ++j2) {
            int colb = (jbase + j2) * 16 + (lane & 15);
            int rb   = mtile * 16 + (lane >> 4) * 8;
#pragma unroll
            for (int i = 0; i < 8; ++i)
                eps[(rb + i) * EPS_STRIDE + colb] = acc[j2][i];
        }
        __syncthreads();

        if (tid < 64) {  // one thread per row: softmax(q), softmax(k)·v/64
            const float* q = &eps[tid * EPS_STRIDE + 0];
            const float* k = &eps[tid * EPS_STRIDE + 64];
            const float* v = &eps[tid * EPS_STRIDE + 128];
            float qmax = -1e30f, kmax = -1e30f;
            for (int d = 0; d < DH; ++d) {
                qmax = fmaxf(qmax, q[d]);
                kmax = fmaxf(kmax, k[d]);
            }
            float qsum = 0.f, ksum = 0.f, kv = 0.f;
            for (int d = 0; d < DH; ++d) {
                qsum += __expf(q[d] - qmax);
                float ek = __expf(k[d] - kmax);
                ksum += ek;
                kv   += ek * v[d];
            }
            float qscale = (kv / (ksum * (float)DH)) / qsum;
            __bf16* op = mid + (size_t)(rowbase + tid) * HIDDEN + h * DH;
            for (int d = 0; d < DH; ++d)
                op[d] = (__bf16)(qscale * __expf(q[d] - qmax));
        }
    }
}

// ---- GEMM2: y[B,2048] = mid[B,512](bf16) * w_out^T + bias ----
__global__ __launch_bounds__(256, 1)
void la_out_kernel(const __bf16* __restrict__ mid,
                   const __bf16* __restrict__ wout_sw,
                   const float* __restrict__ bias,
                   float* __restrict__ y) {
    __shared__ __bf16 ms[128 * MS_STRIDE];  // 130 KB mid tile

    const int tid  = threadIdx.x;
    const int lane = tid & 31;
    const int wave = tid >> 5;
    const int rowbase = (blockIdx.x & 255) * 128;
    const int nb      = blockIdx.x >> 8;      // 16 N-blocks of 128 cols

    // mid tile is already bf16 in the target layout: pure copy -> use CDNA5
    // async global->LDS (no VGPR round trip; tracked by ASYNCcnt).
    for (int t = tid; t < 128 * (HIDDEN / 8); t += 256) {
        int r  = t >> 6;
        int c8 = t & 63;
        unsigned lds_addr = (unsigned)(uintptr_t)(&ms[r * MS_STRIDE + c8 * 8]);
        int voff = (((rowbase + r) * HIDDEN + c8 * 8) * 2);  // byte offset
        async_copy_b128(lds_addr, (const void*)mid, voff);
    }
    wait_async_zero();
    __syncthreads();

    v8f acc[8] = {};
    for (int kb = 0; kb < KT2; ++kb) {
        v16bf a = load_a_frag(ms, MS_STRIDE, wave * 16, kb * 32, lane);
#pragma unroll
        for (int n = 0; n < 8; ++n) {
            v16bf b = load_b_frag(wout_sw, nb * 8 + n, kb, KT2, lane);
            acc[n] = __builtin_amdgcn_wmma_f32_16x16x32_bf16(
                false, a, false, b, (short)0, acc[n], false, false);
        }
    }

    const int rb = rowbase + wave * 16 + (lane >> 4) * 8;
#pragma unroll
    for (int n = 0; n < 8; ++n) {
        int col = nb * 128 + n * 16 + (lane & 15);
        float bv = bias[col];
#pragma unroll
        for (int i = 0; i < 8; ++i)
            y[(size_t)(rb + i) * DIM + col] = acc[n][i] + bv;
    }
}

extern "C" void kernel_launch(void* const* d_in, const int* in_sizes, int n_in,
                              void* d_out, int out_size, void* d_ws, size_t ws_size,
                              hipStream_t stream) {
    const float* x    = (const float*)d_in[0];   // [32768, 2048]
    const float* wqkv = (const float*)d_in[1];   // [1536, 2048]
    const float* wout = (const float*)d_in[2];   // [2048, 512]
    const float* bout = (const float*)d_in[3];   // [2048]
    float* y = (float*)d_out;

    char* ws = (char*)d_ws;
    __bf16* wqkv_sw = (__bf16*)(ws);                           // 6 MiB
    __bf16* wout_sw = (__bf16*)(ws + (size_t)6 * 1024 * 1024); // 2 MiB
    __bf16* mid     = (__bf16*)(ws + (size_t)8 * 1024 * 1024); // 32 MiB

    const int fr1 = NT1 * KT1;  // 6144 fragments
    const int fr2 = NT2 * KT2;  // 2048 fragments
    la_swizzle_w_kernel<<<(fr1 * 32 + 255) / 256, 256, 0, stream>>>(
        wqkv, wqkv_sw, KT1, DIM, fr1);
    la_swizzle_w_kernel<<<(fr2 * 32 + 255) / 256, 256, 0, stream>>>(
        wout, wout_sw, KT2, HIDDEN, fr2);
    la_qkv_attn_kernel<<<B_ROWS / 64, 256, 0, stream>>>(x, wqkv_sw, mid);
    la_out_kernel<<<(B_ROWS / 128) * (DIM / 128), 256, 0, stream>>>(
        mid, wout_sw, bout, y);
}
